// MIA_7000796693177
// MI455X (gfx1250) — matrix operand order repaired and checked by reading, not compiled, not run on hardware
//
#include <hip/hip_runtime.h>
#include <math.h>

#define NUM_USERS 100000
#define NUM_ITEMS 50000
#define NTOT      (NUM_USERS + NUM_ITEMS)
#define D         64
#define Q         128
#define QP        (Q + 8)      // padded LDS column pitch (halves): 272B -> bank stride 4
#define NNZ_CT    3000000
#define NLAYERS   3
#define BATCH     8192
#define OUTC      642

typedef __attribute__((ext_vector_type(16))) _Float16 v16h;
typedef __attribute__((ext_vector_type(8)))  _Float16 v8h;
typedef __attribute__((ext_vector_type(8)))  float    v8f;

__device__ __forceinline__ void atomic_add_f32(float* p, float v) {
    // lowers to global_atomic_add_f32 (native L2 float atomic)
    unsafeAtomicAdd(p, v);
}

__device__ __forceinline__ float sigmoidf_(float x) {
    return 1.0f / (1.0f + expf(-x));
}
__device__ __forceinline__ float softplusf_(float x) {
    // numerically stable softplus
    return fmaxf(x, 0.0f) + log1pf(expf(-fabsf(x)));
}

// ---------------------------------------------------------------------------
// 1) Init: X = A = concat(user_pref, item_pref); Y = 0; zero loss accumulators
// ---------------------------------------------------------------------------
__global__ void k_init(const float* __restrict__ upref, const float* __restrict__ ipref,
                       float* __restrict__ A, float* __restrict__ X, float* __restrict__ Y,
                       float* __restrict__ loss) {
    long long i = (long long)blockIdx.x * blockDim.x + threadIdx.x;
    if (i == 0) { loss[0] = 0.0f; loss[1] = 0.0f; }
    const long long total = (long long)NTOT * D;
    if (i >= total) return;
    const long long usz = (long long)NUM_USERS * D;
    float v = (i < usz) ? upref[i] : ipref[i - usz];
    A[i] = v; X[i] = v; Y[i] = 0.0f;
}

// ---------------------------------------------------------------------------
// 2) SpMM scatter layer: Y[rows[e],:] += vals[e] * X[cols[e],:]
//    One wave32 per edge; each lane owns 2 of the 64 feature dims (float2).
//    38.4MB feature matrix is L2-resident -> atomics run at L2 rate.
// ---------------------------------------------------------------------------
__global__ void k_spmm(const int* __restrict__ rows, const int* __restrict__ cols,
                       const float* __restrict__ vals, const float* __restrict__ X,
                       float* __restrict__ Y) {
    long long t = (long long)blockIdx.x * blockDim.x + threadIdx.x;
    int e    = (int)(t >> 5);
    int lane = (int)(t & 31);
    if (e >= NNZ_CT) return;
    int   r = rows[e];
    int   c = cols[e];
    float v = vals[e];
    const float2 x2 = ((const float2*)(X + (long long)c * D))[lane];
    float* yp = Y + (long long)r * D + lane * 2;
    atomic_add_f32(yp + 0, v * x2.x);
    atomic_add_f32(yp + 1, v * x2.y);
}

// ---------------------------------------------------------------------------
// 3) A += Y; optionally zero Z (old X becomes next layer's Y)
// ---------------------------------------------------------------------------
__global__ void k_accum(float* __restrict__ A, const float* __restrict__ Y,
                        float* __restrict__ Z, int do_zero) {
    long long i = (long long)blockIdx.x * blockDim.x + threadIdx.x;
    const long long total = (long long)NTOT * D;
    if (i >= total) return;
    A[i] += Y[i];
    if (do_zero) Z[i] = 0.0f;
}

// ---------------------------------------------------------------------------
// 4) Gathered GEMM with WMMA: Ob[m,:] = S[idx[m*stride+off], :] @ Mp  (K=Q=128)
//    fp32 emulated via f16 hi/lo split: AB ~= Ahi*Bhi + Alo*Bhi + Ahi*Blo,
//    accumulated in f32 by v_wmma_f32_16x16x32_f16.
//    Block = 128 threads = 4 waves; each wave computes a 16x64 output tile.
//    B kept TRANSPOSED in LDS ([col][k], padded pitch) so each B fragment is
//    two contiguous ds_load_b128 per lane instead of 16 ds_load_u16.
// ---------------------------------------------------------------------------
__global__ __launch_bounds__(128)
void k_gemm_gather(const int* __restrict__ idx, int idx_stride, int idx_off,
                   const float* __restrict__ S,   // [*, Q]
                   const float* __restrict__ Mp,  // [Q, D] row-major
                   float* __restrict__ Ob) {      // [BATCH, D]
    __shared__ _Float16 sBhiT[D * QP];   // transposed: [col][k]
    __shared__ _Float16 sBloT[D * QP];

    const int lane = threadIdx.x & 31;
    const int wave = threadIdx.x >> 5;
    const int hi16 = lane >> 4;
    const int m0   = (blockIdx.x * 4 + wave) * 16;
    const int mrow = m0 + (lane & 15);
    const int gidx = idx[mrow * idx_stride + idx_off];
    const float* aRow = S + (long long)gidx * Q;

    // start pulling the gathered row toward the caches while we convert B
    __builtin_prefetch(aRow, 0, 3);                 // global_prefetch_b8
    __builtin_prefetch(aRow + 64, 0, 3);

    for (int j = threadIdx.x; j < Q * D; j += blockDim.x) {
        const int k   = j >> 6;        // row of Mp (K index)
        const int col = j & 63;        // col of Mp (N index)
        float v = Mp[j];
        _Float16 h = (_Float16)v;
        sBhiT[col * QP + k] = h;
        sBloT[col * QP + k] = (_Float16)(v - (float)h);
    }
    __syncthreads();

    v8f acc[4] = {};

    #pragma unroll
    for (int kt = 0; kt < Q; kt += 32) {
        // A tile 16x32: lane holds row M=lane%16; halves 0..7 -> K=kbase+i,
        // halves 8..15 -> K=kbase+16+i  (kbase = kt + (lane/16)*8)
        const int kbase = kt + hi16 * 8;
        const float4 a0 = *(const float4*)(aRow + kbase);
        const float4 a1 = *(const float4*)(aRow + kbase + 4);
        const float4 a2 = *(const float4*)(aRow + kbase + 16);
        const float4 a3 = *(const float4*)(aRow + kbase + 20);
        const float af[16] = { a0.x, a0.y, a0.z, a0.w,  a1.x, a1.y, a1.z, a1.w,
                               a2.x, a2.y, a2.z, a2.w,  a3.x, a3.y, a3.z, a3.w };
        v16h ahi, alo;
        #pragma unroll
        for (int i = 0; i < 16; ++i) {
            _Float16 h = (_Float16)af[i];
            ahi[i] = h;
            alo[i] = (_Float16)(af[i] - (float)h);
        }
        // B tile 32x16: lane holds col N=lane%16; halves 0..7 -> K=koff+i,
        // halves 8..15 -> K=koff+16+i  (koff = kt + (lane/16)*8)
        const int koff = kt + hi16 * 8;
        #pragma unroll
        for (int nt = 0; nt < 4; ++nt) {
            const int col  = nt * 16 + (lane & 15);
            const int base = col * QP + koff;
            const v8h bh0 = *(const v8h*)(sBhiT + base);
            const v8h bh1 = *(const v8h*)(sBhiT + base + 16);
            const v8h bl0 = *(const v8h*)(sBloT + base);
            const v8h bl1 = *(const v8h*)(sBloT + base + 16);
            const v16h bhi = __builtin_shufflevector(bh0, bh1,
                0,1,2,3,4,5,6,7,8,9,10,11,12,13,14,15);
            const v16h blo = __builtin_shufflevector(bl0, bl1,
                0,1,2,3,4,5,6,7,8,9,10,11,12,13,14,15);
            acc[nt] = __builtin_amdgcn_wmma_f32_16x16x32_f16(
                false, ahi, false, bhi, (short)0, acc[nt], false, false);
            acc[nt] = __builtin_amdgcn_wmma_f32_16x16x32_f16(
                false, alo, false, bhi, (short)0, acc[nt], false, false);
            acc[nt] = __builtin_amdgcn_wmma_f32_16x16x32_f16(
                false, ahi, false, blo, (short)0, acc[nt], false, false);
        }
    }

    // C/D layout: VGPR r -> M = (lane/16)*8 + r, N = lane%16
    #pragma unroll
    for (int nt = 0; nt < 4; ++nt) {
        const int col = nt * 16 + (lane & 15);
        #pragma unroll
        for (int r = 0; r < 8; ++r) {
            int m = m0 + hi16 * 8 + r;
            Ob[(long long)m * D + col] = acc[nt][r];
        }
    }
}

// ---------------------------------------------------------------------------
// 5) Batch epilogue: wave32 per batch element. 6 dot products via shfl_xor
//    butterflies, weight update + strong-item selection, loss accumulation,
//    and output columns [0, 640).
// ---------------------------------------------------------------------------
__global__ __launch_bounds__(256)
void k_batch(const float* __restrict__ A,
             const float* __restrict__ USb, const float* __restrict__ ASb,
             const float* __restrict__ WSb, const float* __restrict__ SSb,
             const int* __restrict__ users, const int* __restrict__ adj,
             const int* __restrict__ pool, const float* __restrict__ pwt,
             const float* __restrict__ udec, const float* __restrict__ idec,
             float* __restrict__ out, float* __restrict__ loss) {
    const int i    = (int)(((long long)blockIdx.x * blockDim.x + threadIdx.x) >> 5);
    const int lane = threadIdx.x & 31;
    if (i >= BATCH) return;

    const int u  = users[i];
    const int ai = adj[i];
    const int wi = pool[2 * i + 0];
    const int si = pool[2 * i + 1];

    const float2 up = ((const float2*)(A + (long long)u * D))[lane];
    const float2 ap = ((const float2*)(A + (long long)(NUM_USERS + ai) * D))[lane];
    const float2 wp = ((const float2*)(A + (long long)(NUM_USERS + wi) * D))[lane];
    const float2 sp = ((const float2*)(A + (long long)(NUM_USERS + si) * D))[lane];
    const float2 us = ((const float2*)(USb + (long long)i * D))[lane];
    const float2 as = ((const float2*)(ASb + (long long)i * D))[lane];
    const float2 ws = ((const float2*)(WSb + (long long)i * D))[lane];
    const float2 ss = ((const float2*)(SSb + (long long)i * D))[lane];

    float d0 = up.x * ap.x + up.y * ap.y;  // u_p . a_p
    float d1 = up.x * wp.x + up.y * wp.y;  // u_p . w_p
    float d2 = up.x * sp.x + up.y * sp.y;  // u_p . s_p
    float d3 = us.x * as.x + us.y * as.y;  // u_s . a_s (adj_sc)
    float d4 = us.x * ws.x + us.y * ws.y;  // u_s . w_s (weak_sc)
    float d5 = us.x * ss.x + us.y * ss.y;  // u_s . s_s (strong_sc)

    #pragma unroll
    for (int m = 16; m >= 1; m >>= 1) {
        d0 += __shfl_xor(d0, m, 32);
        d1 += __shfl_xor(d1, m, 32);
        d2 += __shfl_xor(d2, m, 32);
        d3 += __shfl_xor(d3, m, 32);
        d4 += __shfl_xor(d4, m, 32);
        d5 += __shfl_xor(d5, m, 32);
    }

    const float weak_w   = sigmoidf_(log1pf(pwt[2 * i + 0])) + sigmoidf_(d1 + d4);
    const float strong_w = sigmoidf_(log1pf(pwt[2 * i + 1])) + sigmoidf_(d2 + d5);
    const bool  take_weak = (strong_w < weak_w);

    if (lane == 0) {
        float pl = softplusf_(-d0);  // -log_sigmoid(u_p . a_p)
        float sl = 0.5f * (softplusf_(d5 - d3) + softplusf_(d4 - d5));
        atomic_add_f32(&loss[0], pl);
        atomic_add_f32(&loss[1], sl);
    }

    const float2 sp2 = take_weak ? wp : sp;
    const float2 ss2 = take_weak ? ws : ss;
    const float2 ud  = ((const float2*)(udec + (long long)u * D))[lane];
    const float2 id  = ((const float2*)(idec + (long long)ai * D))[lane];

    float* o = out + (long long)i * OUTC;
    ((float2*)(o +   0))[lane] = up;   // users_exposure: u_p
    ((float2*)(o +  64))[lane] = us;   //                 u_s
    ((float2*)(o + 128))[lane] = ap;   // adj_exposure:   a_p
    ((float2*)(o + 192))[lane] = as;   //                 a_s
    ((float2*)(o + 256))[lane] = sp2;  // strong_exposure: items_pref[update_strong]
    ((float2*)(o + 320))[lane] = ss2;  //                  items_structure[update_strong]
    ((float2*)(o + 384))[lane] = up;   // users_decision_e: u_p
    ((float2*)(o + 448))[lane] = ud;   //                   user_decision[u]
    ((float2*)(o + 512))[lane] = ap;   // adj_decision_e:   a_p
    ((float2*)(o + 576))[lane] = id;   //                   item_decision[ai]
}

// ---------------------------------------------------------------------------
// 6) Broadcast mean losses into columns 640/641 of every output row
// ---------------------------------------------------------------------------
__global__ void k_loss(const float* __restrict__ loss, float* __restrict__ out) {
    int i = blockIdx.x * blockDim.x + threadIdx.x;
    if (i >= BATCH) return;
    const float inv = 1.0f / (float)BATCH;
    out[(long long)i * OUTC + 640] = loss[0] * inv;
    out[(long long)i * OUTC + 641] = loss[1] * inv;
}

// ---------------------------------------------------------------------------
extern "C" void kernel_launch(void* const* d_in, const int* in_sizes, int n_in,
                              void* d_out, int out_size, void* d_ws, size_t ws_size,
                              hipStream_t stream) {
    (void)in_sizes; (void)n_in; (void)out_size; (void)ws_size;

    const int*   rows  = (const int*)d_in[0];
    const int*   cols  = (const int*)d_in[1];
    const float* vals  = (const float*)d_in[2];
    const float* upref = (const float*)d_in[3];
    const float* ipref = (const float*)d_in[4];
    const float* UmS   = (const float*)d_in[5];
    const float* VmS   = (const float*)d_in[6];
    const float* umap  = (const float*)d_in[7];
    const float* imap  = (const float*)d_in[8];
    const float* udec  = (const float*)d_in[9];
    const float* idec  = (const float*)d_in[10];
    const int*   users = (const int*)d_in[11];
    const int*   adj   = (const int*)d_in[12];
    const int*   pool  = (const int*)d_in[13];
    const float* pwt   = (const float*)d_in[14];
    float* out = (float*)d_out;

    // workspace carve-up
    const long long nd = (long long)NTOT * D;
    float* ws   = (float*)d_ws;
    float* A    = ws;                             // acc   [N, D]
    float* X    = A + nd;                         // x     [N, D]
    float* Y    = X + nd;                         // layer out [N, D]
    float* USb  = Y + nd;                         // [B, D]
    float* ASb  = USb + (long long)BATCH * D;
    float* WSb  = ASb + (long long)BATCH * D;
    float* SSb  = WSb + (long long)BATCH * D;
    float* loss = SSb + (long long)BATCH * D;     // 2 floats

    const int TB = 256;

    const int initBlocks = (int)((nd + TB - 1) / TB);
    k_init<<<initBlocks, TB, 0, stream>>>(upref, ipref, A, X, Y, loss);

    // batch-gathered structure GEMMs (WMMA)
    const dim3 gg(BATCH / 64);
    k_gemm_gather<<<gg, 128, 0, stream>>>(users, 1, 0, UmS, umap, USb);
    k_gemm_gather<<<gg, 128, 0, stream>>>(adj,   1, 0, VmS, imap, ASb);
    k_gemm_gather<<<gg, 128, 0, stream>>>(pool,  2, 0, VmS, imap, WSb);
    k_gemm_gather<<<gg, 128, 0, stream>>>(pool,  2, 1, VmS, imap, SSb);

    // 3-layer propagation GCN
    const long long spmmThreads = (long long)NNZ_CT * 32;
    const int spmmBlocks = (int)((spmmThreads + TB - 1) / TB);
    float* Xc = X;
    float* Yc = Y;
    for (int l = 0; l < NLAYERS; ++l) {
        k_spmm<<<spmmBlocks, TB, 0, stream>>>(rows, cols, vals, Xc, Yc);
        k_accum<<<initBlocks, TB, 0, stream>>>(A, Yc, Xc, (l < NLAYERS - 1) ? 1 : 0);
        float* t = Xc; Xc = Yc; Yc = t;
    }

    // epilogue
    k_batch<<<(BATCH * 32) / TB, TB, 0, stream>>>(A, USb, ASb, WSb, SSb,
                                                  users, adj, pool, pwt,
                                                  udec, idec, out, loss);
    k_loss<<<(BATCH + TB - 1) / TB, TB, 0, stream>>>(loss, out);
}
